// GraphAttentionLayer_56504589746809
// MI455X (gfx1250) — matrix operand order repaired and checked by reading, not compiled, not run on hardware
//
#include <hip/hip_runtime.h>
#include <hip/hip_bf16.h>
#include <math.h>

#define NTOT 8192
#define DIN  512
#define DOUT 256
#define LRELU_ALPHA 0.2f
#define NEG_INF_F (-9e15f)

typedef __attribute__((ext_vector_type(2)))  float    v2f;
typedef __attribute__((ext_vector_type(8)))  float    v8f;
typedef __attribute__((ext_vector_type(8)))  _Float16 v8h;
typedef __attribute__((ext_vector_type(16))) _Float16 v16h;
typedef __attribute__((ext_vector_type(4)))  int      v4i;

// ---- CDNA5 async global->LDS path (ASYNCcnt), with portable fallback ------
#if defined(__gfx1250__) && __has_builtin(__builtin_amdgcn_global_load_async_to_lds_b128)
#define HAVE_ASYNC_LDS 1
typedef __attribute__((address_space(1))) v4i* gbl_v4i_p;
typedef __attribute__((address_space(3))) v4i* lds_v4i_p;
#else
#define HAVE_ASYNC_LDS 0
#endif

__device__ __forceinline__ void async_copy16(const void* g, void* l) {
#if HAVE_ASYNC_LDS
  __builtin_amdgcn_global_load_async_to_lds_b128((gbl_v4i_p)g, (lds_v4i_p)l, 0, 0);
#else
  *(int4*)l = *(const int4*)g;
#endif
}

__device__ __forceinline__ void wait_async0() {
#if HAVE_ASYNC_LDS
#if __has_builtin(__builtin_amdgcn_s_wait_asynccnt)
  __builtin_amdgcn_s_wait_asynccnt(0);
#else
  asm volatile("s_wait_asynccnt 0x0" ::: "memory");
#endif
#endif
}

// ---------------------------------------------------------------------------
// Kernel 1: Wh = X @ W   [8192,512]@[512,256], fp32 WMMA 16x16x4 (full fp32
// because Wh feeds the softmax logits). block = 128 (4 waves), 16-row tile.
// ---------------------------------------------------------------------------
__global__ __launch_bounds__(128)
void wh_kernel(const float* __restrict__ X, const float* __restrict__ W,
               float* __restrict__ Wh) {
  __shared__ __align__(16) float sX[16][DIN + 4];  // stride 516: conflict-free
  const int tid = threadIdx.x;
  const int r0  = blockIdx.x * 16;

  for (int t = 0; t < 16; ++t) {
    int idx = tid + t * 128;            // 2048 float4s
    int row = idx >> 7;
    int c4  = (idx & 127) << 2;
    async_copy16(X + (r0 + row) * DIN + c4, &sX[row][c4]);
  }
  wait_async0();
  __syncthreads();

  const int wave = tid >> 5;
  const int lane = tid & 31;
  const int half = lane >> 4;
  const int l16  = lane & 15;

  v8f acc[4] = {};
  const int nbase = wave * 64;
  for (int kt = 0; kt < DIN / 4; ++kt) {
    const int k0 = kt * 4 + 2 * half;   // A-frag: lanes<16 K=0,1; lanes>=16 K=2,3
    v2f a;
    a.x = sX[l16][k0 + 0];
    a.y = sX[l16][k0 + 1];
#pragma unroll
    for (int j = 0; j < 4; ++j) {
      const int n = nbase + j * 16 + l16;
      v2f b;
      b.x = W[(k0 + 0) * DOUT + n];
      b.y = W[(k0 + 1) * DOUT + n];
      acc[j] = __builtin_amdgcn_wmma_f32_16x16x4_f32(false, a, false, b,
                                                     (short)0, acc[j], false, false);
    }
  }
#pragma unroll
  for (int j = 0; j < 4; ++j) {
    const int n = nbase + j * 16 + l16;
#pragma unroll
    for (int i = 0; i < 8; ++i)
      Wh[(r0 + i + 8 * half) * DOUT + n] = acc[j][i];
  }
}

// ---------------------------------------------------------------------------
// Kernel 2: src/dst GEMV (wave per row, shuffle reduction), fp32.
// ---------------------------------------------------------------------------
__global__ __launch_bounds__(256)
void src_dst_kernel(const float* __restrict__ Wh, const float* __restrict__ av,
                    float* __restrict__ srcv, float* __restrict__ dstv) {
  const int wave = threadIdx.x >> 5;
  const int lane = threadIdx.x & 31;
  const int row  = blockIdx.x * 8 + wave;
  float s = 0.f, d = 0.f;
  for (int n = lane; n < DOUT; n += 32) {
    const float w = Wh[row * DOUT + n];
    s += w * av[n];
    d += w * av[DOUT + n];
  }
#pragma unroll
  for (int m = 16; m >= 1; m >>= 1) {
    s += __shfl_xor(s, m, 32);
    d += __shfl_xor(d, m, 32);
  }
  if (lane == 0) { srcv[row] = s; dstv[row] = d; }
}

// ---------------------------------------------------------------------------
// Kernel 3: WhT16[n][k] = (f16) Wh[k][n]  (64x64 LDS-tiled transpose).
// Gives contiguous-K f16 B-fragments for the 16x16x32 WMMA.
// ---------------------------------------------------------------------------
__global__ __launch_bounds__(256)
void wh_to_f16T_kernel(const float* __restrict__ Wh, _Float16* __restrict__ WhT) {
  __shared__ _Float16 sT[64][68];
  const int tid = threadIdx.x;
  const int k0  = blockIdx.x * 64;   // over NTOT rows of Wh
  const int n0  = blockIdx.y * 64;   // over DOUT cols of Wh
#pragma unroll 4
  for (int t = 0; t < 16; ++t) {     // coalesced read along n
    const int kk = (tid >> 6) + t * 4;
    const int nn = tid & 63;
    sT[kk][nn] = (_Float16)Wh[(k0 + kk) * DOUT + n0 + nn];
  }
  __syncthreads();
#pragma unroll 4
  for (int t = 0; t < 16; ++t) {     // coalesced write along k
    const int nn = (tid >> 6) + t * 4;
    const int kk = tid & 63;
    WhT[(size_t)(n0 + nn) * NTOT + k0 + kk] = sT[kk][nn];
  }
}

// ---------------------------------------------------------------------------
// Kernel 4: fused masked-softmax + P @ Wh + ELU, flash style.
// Scores/softmax/accumulators fp32; value matmul via v_wmma_f32_16x16x32_f16.
// adj tiles double-buffered through LDS with async global->LDS loads.
// block = 256 (8 waves); wave owns output cols [w*32, w*32+32).
// ---------------------------------------------------------------------------
__global__ __launch_bounds__(256)
void attn_kernel(const int* __restrict__ adj, const _Float16* __restrict__ WhT,
                 const float* __restrict__ srcv, const float* __restrict__ dstv,
                 float* __restrict__ out) {
  __shared__ __align__(16) int      sAdj[2][16][64];   // double-buffered adj tile
  __shared__ __align__(16) _Float16 sP[16][80];        // f16 probs, 16B-aligned frags
  __shared__ float sM[16], sL[16], sScale[16];

  const int tid  = threadIdx.x;
  const int r0   = blockIdx.x * 16;
  const int r    = tid >> 4;          // score row 0..15
  const int c16  = tid & 15;          // score col group
  const int wave = tid >> 5;
  const int lane = tid & 31;
  const int half = lane >> 4;
  const int l16  = lane & 15;

  if (tid < 16) { sM[tid] = -INFINITY; sL[tid] = 0.f; }
  v8f acc[2] = {};
  const float srcr = srcv[r0 + r];

  // prologue: async-load tile 0
  async_copy16(adj + (size_t)(r0 + r) * NTOT + c16 * 4, &sAdj[0][r][c16 * 4]);
  wait_async0();
  __syncthreads();

  for (int ti = 0; ti < NTOT / 64; ++ti) {
    const int buf = ti & 1;
    if (ti + 1 < NTOT / 64)   // overlap next adj tile with this tile's work
      async_copy16(adj + (size_t)(r0 + r) * NTOT + (ti + 1) * 64 + c16 * 4,
                   &sAdj[buf ^ 1][r][c16 * 4]);

    // ---- masked leaky-relu scores (4 per thread) ----
    const int cbase = ti * 64 + c16 * 4;
    float e[4];
#pragma unroll
    for (int j = 0; j < 4; ++j) {
      float s = srcr + dstv[cbase + j];
      s = s > 0.f ? s : LRELU_ALPHA * s;
      e[j] = sAdj[buf][r][c16 * 4 + j] > 0 ? s : NEG_INF_F;
    }

    // ---- online softmax: row max over the 16 threads of this row ----
    float tmax = fmaxf(fmaxf(e[0], e[1]), fmaxf(e[2], e[3]));
#pragma unroll
    for (int m = 8; m >= 1; m >>= 1)
      tmax = fmaxf(tmax, __shfl_xor(tmax, m, 16));
    if (c16 == 0) {
      const float mold = sM[r];
      const float mnew = fmaxf(mold, tmax);
      sScale[r] = __expf(mold - mnew);
      sM[r] = mnew;
    }
    __syncthreads();

    const float mnew = sM[r];
    float tsum = 0.f;
#pragma unroll
    for (int j = 0; j < 4; ++j) {
      const float p = __expf(e[j] - mnew);
      sP[r][c16 * 4 + j] = (_Float16)p;   // f16 for the WMMA A-matrix
      tsum += p;
    }
#pragma unroll
    for (int m = 8; m >= 1; m >>= 1)
      tsum += __shfl_xor(tsum, m, 16);
    if (c16 == 0) sL[r] = sL[r] * sScale[r] + tsum;
    __syncthreads();

    // ---- rescale fp32 accumulators by exp(m_old - m_new) ----
#pragma unroll
    for (int i = 0; i < 8; ++i) {
      const float f = sScale[i + 8 * half];
      acc[0][i] *= f;
      acc[1][i] *= f;
    }

    // ---- P(16x64) @ Wh[tile,:] via f16 WMMA (two K=32 windows) ----
#pragma unroll
    for (int kb = 0; kb < 64; kb += 32) {
      // A-frag: elem i = K kb+16*half+i (contiguous, matches B)
      const v16h a = *(const v16h*)&sP[l16][kb + 16 * half];
#pragma unroll
      for (int t = 0; t < 2; ++t) {
        const int n = wave * 32 + t * 16 + l16;
        // B-frag: elem i = Wh[K = tile*64 + kb + 16*half + i][n], contiguous in WhT
        const _Float16* bp = WhT + (size_t)n * NTOT + ti * 64 + kb + 16 * half;
        if (ti + 1 < NTOT / 64)
          __builtin_prefetch(bp + 64, 0, 1);   // global_prefetch_b8 next tile
        const v16h b = *(const v16h*)bp;
        acc[t] = __builtin_amdgcn_wmma_f32_16x16x32_f16(false, a, false, b,
                                                        (short)0, acc[t], false, false);
      }
    }
    wait_async0();       // next adj tile landed in LDS
    __syncthreads();     // sP/sScale reused next iteration
  }

  // ---- finalize: 1/l softmax denominator + ELU ----
#pragma unroll
  for (int i = 0; i < 8; ++i) {
    const int row = i + 8 * half;
    const float invl = 1.0f / sL[row];
#pragma unroll
    for (int t = 0; t < 2; ++t) {
      const int n = wave * 32 + t * 16 + l16;
      const float h = acc[t][i] * invl;
      out[(r0 + row) * DOUT + n] = h > 0.f ? h : (__expf(h) - 1.f);
    }
  }
}

// ---------------------------------------------------------------------------
extern "C" void kernel_launch(void* const* d_in, const int* in_sizes, int n_in,
                              void* d_out, int out_size, void* d_ws, size_t ws_size,
                              hipStream_t stream) {
  const float* X   = (const float*)d_in[0];   // [8192,512]
  const int*   adj = (const int*)d_in[1];     // [8192,8192]
  const float* W   = (const float*)d_in[2];   // [512,256]
  const float* a   = (const float*)d_in[3];   // [512,1]
  float* out = (float*)d_out;                 // [8192,256]

  float* ws   = (float*)d_ws;
  float* Wh   = ws;                               // 8192*256 fp32 (8 MB)
  float* srcv = ws + (size_t)NTOT * DOUT;         // 8192 fp32
  float* dstv = srcv + NTOT;                      // 8192 fp32
  _Float16* WhT = (_Float16*)(dstv + NTOT);       // 256*8192 f16 (4 MB)

  wh_kernel<<<NTOT / 16, 128, 0, stream>>>(X, W, Wh);
  src_dst_kernel<<<NTOT / 8, 256, 0, stream>>>(Wh, a, srcv, dstv);
  wh_to_f16T_kernel<<<dim3(NTOT / 64, DOUT / 64), 256, 0, stream>>>(Wh, WhT);
  attn_kernel<<<NTOT / 16, 256, 0, stream>>>(adj, WhT, srcv, dstv, out);
}
